// DGN_66348654788984
// MI455X (gfx1250) — matrix operand order repaired and checked by reading, not compiled, not run on hardware
//
#include <hip/hip_runtime.h>
#include <hip/hip_bf16.h>
#include <math.h>

#define V_SZ 50257
#define E_SZ 512
#define H_SZ 512
#define C_SZ 64
#define F_SZ 32
#define L_SZ 64
#define K2_SZ 1024   // 2*H

typedef __bf16 bf16_t;
typedef __attribute__((ext_vector_type(16))) __bf16 v16bf;
typedef __attribute__((ext_vector_type(8)))  __bf16 v8bf;
typedef __attribute__((ext_vector_type(8)))  float  v8f;

// ---------------------------------------------------------------------------
// Setup: facts[c][e] = sum_s emb_ctx[context[c,s]][e] * l(s,e);  also zero nll
// ---------------------------------------------------------------------------
__global__ void facts_kernel(const int* __restrict__ context,
                             const float* __restrict__ emb_ctx,
                             float* __restrict__ facts,
                             float* __restrict__ nll_acc) {
    const int c = blockIdx.x;
    const int tid = threadIdx.x;
    if (c == 0 && tid == 0) *nll_acc = 0.0f;
    for (int e = tid; e < E_SZ; e += blockDim.x) {
        const float ef = (float)e / (float)(E_SZ - 1);
        float acc = 0.0f;
        for (int s = 0; s < F_SZ; ++s) {
            const float sf = (float)s / (float)(F_SZ - 1);
            const float l = 1.0f - sf - ef * (1.0f - 2.0f * sf);
            const int tok = context[c * F_SZ + s];
            acc += emb_ctx[(size_t)tok * E_SZ + e] * l;
        }
        facts[c * E_SZ + e] = acc;
    }
}

// ---------------------------------------------------------------------------
// Setup: pack W1 [H, 2H] (row-major, f32) into bf16 WMMA B-fragment layout.
// B element (k, n) = W1[n][k].  For tile (kt, nt) and lane L:
//   n = nt*16 + (L%16), kbase = kt*32 + (L/16)*16, 16 contiguous K values.
// ---------------------------------------------------------------------------
__global__ void __launch_bounds__(32) packW1_kernel(const float* __restrict__ W1,
                                                    bf16_t* __restrict__ Bp) {
    const int b = blockIdx.x;          // b = kt*32 + nt,  kt,nt in 0..31
    const int kt = b >> 5;
    const int nt = b & 31;
    const int L = threadIdx.x;
    const int n = nt * 16 + (L & 15);
    const int kb = kt * 32 + (L >> 4) * 16;
    bf16_t* dst = Bp + ((size_t)(b * 32 + L)) * 16;
    for (int kk = 0; kk < 16; ++kk)
        dst[kk] = (bf16_t)W1[(size_t)n * K2_SZ + kb + kk];
}

__device__ inline float block_reduce(float v, float* red, int tid, int n) {
    red[tid] = v; __syncthreads();
    for (int s = n >> 1; s > 0; s >>= 1) {
        if (tid < s) red[tid] += red[tid + s];
        __syncthreads();
    }
    float r = red[0]; __syncthreads();
    return r;
}

// ---------------------------------------------------------------------------
// Setup: m0[j] = relu(Wm[j,:] . facts_flat + bm[j]);  h0[j] = 0
// ---------------------------------------------------------------------------
__global__ void m0_kernel(const float* __restrict__ Wm,
                          const float* __restrict__ bm,
                          const float* __restrict__ facts,
                          float* __restrict__ m0,
                          float* __restrict__ h0) {
    __shared__ float red[256];
    const int j = blockIdx.x;
    const int tid = threadIdx.x;
    const float* row = Wm + (size_t)j * (C_SZ * E_SZ);
    float p = 0.0f;
    for (int i = tid; i < C_SZ * E_SZ; i += 256) p += row[i] * facts[i];
    p = block_reduce(p, red, tid, 256);
    if (tid == 0) {
        float v = p + bm[j];
        m0[j] = v > 0.0f ? v : 0.0f;
        h0[j] = 0.0f;
    }
}

// ---------------------------------------------------------------------------
// GRU gate matvecs: gi[j] = Wih[j,:].[ei;m] + bih[j]; gh[j] = Whh[j,:].h + bhh[j]
// ---------------------------------------------------------------------------
__global__ void gru_matvec_kernel(const float* __restrict__ emb_dec,
                                  const int* __restrict__ description,
                                  const float* __restrict__ Wih,
                                  const float* __restrict__ Whh,
                                  const float* __restrict__ bih,
                                  const float* __restrict__ bhh,
                                  const float* __restrict__ m_old,
                                  const float* __restrict__ h_old,
                                  float* __restrict__ gi,
                                  float* __restrict__ gh,
                                  int t) {
    __shared__ float red1[128];
    __shared__ float red2[128];
    const int j = blockIdx.x;          // 0..1535
    const int tid = threadIdx.x;
    const int tok = (t == 0) ? 1 : description[t - 1];
    const float* ei = emb_dec + (size_t)tok * E_SZ;
    const float* wi = Wih + (size_t)j * (E_SZ + H_SZ);
    const float* wh = Whh + (size_t)j * H_SZ;
    float p1 = 0.0f, p2 = 0.0f;
    for (int k = tid; k < E_SZ; k += 128) {
        p1 += wi[k] * ei[k];
        p1 += wi[E_SZ + k] * m_old[k];
        p2 += wh[k] * h_old[k];
    }
    p1 = block_reduce(p1, red1, tid, 128);
    p2 = block_reduce(p2, red2, tid, 128);
    if (tid == 0) {
        gi[j] = p1 + bih[j];
        gh[j] = p2 + bhh[j];
    }
}

__global__ void gru_combine_kernel(const float* __restrict__ gi,
                                   const float* __restrict__ gh,
                                   const float* __restrict__ h_old,
                                   float* __restrict__ h_new) {
    const int j = blockIdx.x * blockDim.x + threadIdx.x;
    if (j >= H_SZ) return;
    const float r = 1.0f / (1.0f + expf(-(gi[j] + gh[j])));
    const float z = 1.0f / (1.0f + expf(-(gi[H_SZ + j] + gh[H_SZ + j])));
    const float n = tanhf(gi[2 * H_SZ + j] + r * gh[2 * H_SZ + j]);
    h_new[j] = (1.0f - z) * n + z * h_old[j];
}

// ---------------------------------------------------------------------------
// zc[c][j] = |facts[c][j] - h_new[j]| (j<512) else |facts[c][j-512] - m_old[j-512]|
// stored as bf16 (WMMA A operand)
// ---------------------------------------------------------------------------
__global__ void zc_kernel(const float* __restrict__ facts,
                          const float* __restrict__ h_new,
                          const float* __restrict__ m_old,
                          bf16_t* __restrict__ zc) {
    const int c = blockIdx.x;
    for (int j = threadIdx.x; j < K2_SZ; j += blockDim.x) {
        float v;
        if (j < H_SZ) v = fabsf(facts[c * E_SZ + j] - h_new[j]);
        else          v = fabsf(facts[c * E_SZ + (j - H_SZ)] - m_old[j - H_SZ]);
        zc[c * K2_SZ + j] = (bf16_t)v;
    }
}

// ---------------------------------------------------------------------------
// Attention GEMM: a_tanh[64,512] = tanh( zc[64,1024] @ W1^T + b1 )
// One wave per 16x16 output tile; 32 K-steps of v_wmma_f32_16x16x32_bf16.
// ---------------------------------------------------------------------------
__global__ void __launch_bounds__(32) attn_wmma_kernel(const bf16_t* __restrict__ zc,
                                                       const bf16_t* __restrict__ Bp,
                                                       const float* __restrict__ b1,
                                                       float* __restrict__ a_tanh) {
    const int mt = blockIdx.x;          // 0..3   (M tiles, C=64)
    const int nt = blockIdx.y;          // 0..31  (N tiles, H=512)
    const int lane = threadIdx.x;       // 0..31
    const int mrow = mt * 16 + (lane & 15);
    const int kgrp = lane >> 4;         // 0 or 1

    v8f acc = {};
    for (int kt = 0; kt < 32; ++kt) {
        // A fragment: 16 contiguous K values per lane (kbase = kgrp*16), matching packed B
        const bf16_t* arow = zc + (size_t)mrow * K2_SZ + kt * 32 + kgrp * 16;
        v8bf alo = *(const v8bf*)(arow);
        v8bf ahi = *(const v8bf*)(arow + 8);
        v16bf a;
#pragma unroll
        for (int i = 0; i < 8; ++i) { a[i] = alo[i]; a[8 + i] = ahi[i]; }

        // B fragment: pre-packed, 16 contiguous bf16 per lane
        const bf16_t* bp = Bp + ((size_t)((kt * 32 + nt) * 32 + lane)) * 16;
        v8bf blo = *(const v8bf*)(bp);
        v8bf bhi = *(const v8bf*)(bp + 8);
        v16bf b;
#pragma unroll
        for (int i = 0; i < 8; ++i) { b[i] = blo[i]; b[8 + i] = bhi[i]; }

        acc = __builtin_amdgcn_wmma_f32_16x16x32_bf16(
            false, a, false, b, (short)0, acc, false, false);
    }

    const int ncol = nt * 16 + (lane & 15);
    const float bias = b1[ncol];
#pragma unroll
    for (int i = 0; i < 8; ++i) {
        const int row = mt * 16 + kgrp * 8 + i;       // C/D layout: VGPR i -> M
        a_tanh[(size_t)row * H_SZ + ncol] = tanhf(acc[i] + bias);
    }
}

// ---------------------------------------------------------------------------
// scores[c] = a_tanh[c,:].W2 + b2 ; g = softmax(scores) ; c_vec = g @ facts
// ---------------------------------------------------------------------------
__global__ void score_ctx_kernel(const float* __restrict__ a_tanh,
                                 const float* __restrict__ W2,
                                 const float* __restrict__ b2,
                                 const float* __restrict__ facts,
                                 float* __restrict__ c_vec) {
    __shared__ float red[256];
    __shared__ float sc[C_SZ];
    __shared__ float stats[2];          // max, sum
    const int tid = threadIdx.x;
    const int c = tid >> 2;             // 4 threads per row
    const int sub = tid & 3;
    float p = 0.0f;
    for (int n = sub; n < H_SZ; n += 4) p += a_tanh[c * H_SZ + n] * W2[n];
    red[tid] = p; __syncthreads();
    if (sub == 0) sc[c] = red[tid] + red[tid + 1] + red[tid + 2] + red[tid + 3] + b2[0];
    __syncthreads();
    if (tid == 0) {
        float mx = sc[0];
        for (int i = 1; i < C_SZ; ++i) mx = fmaxf(mx, sc[i]);
        float s = 0.0f;
        for (int i = 0; i < C_SZ; ++i) s += expf(sc[i] - mx);
        stats[0] = mx; stats[1] = 1.0f / s;
    }
    __syncthreads();
    if (tid < C_SZ) sc[tid] = expf(sc[tid] - stats[0]) * stats[1];
    __syncthreads();
    for (int h = tid; h < H_SZ; h += 256) {
        float acc = 0.0f;
        for (int i = 0; i < C_SZ; ++i) acc += sc[i] * facts[i * E_SZ + h];
        c_vec[h] = acc;
    }
}

// ---------------------------------------------------------------------------
// m_new[j] = relu(W3[j,:].[m;c;h] + b3[j]);  out[j] = tanh(Wo1[j,:].[h;c] + bo1[j])
// blocks 0..511 -> m_new, 512..1023 -> out
// ---------------------------------------------------------------------------
__global__ void memout_kernel(const float* __restrict__ W3,
                              const float* __restrict__ b3,
                              const float* __restrict__ Wo1,
                              const float* __restrict__ bo1,
                              const float* __restrict__ m_old,
                              const float* __restrict__ c_vec,
                              const float* __restrict__ h_new,
                              float* __restrict__ m_new,
                              float* __restrict__ out_vec) {
    __shared__ float red[128];
    const int tid = threadIdx.x;
    if (blockIdx.x < 512) {
        const int j = blockIdx.x;
        const float* row = W3 + (size_t)j * (3 * H_SZ);
        float p = 0.0f;
        for (int i = tid; i < 3 * H_SZ; i += 128) {
            float x = (i < H_SZ) ? m_old[i] : (i < 2 * H_SZ) ? c_vec[i - H_SZ]
                                                             : h_new[i - 2 * H_SZ];
            p += row[i] * x;
        }
        p = block_reduce(p, red, tid, 128);
        if (tid == 0) { float v = p + b3[j]; m_new[j] = v > 0.0f ? v : 0.0f; }
    } else {
        const int j = blockIdx.x - 512;
        const float* row = Wo1 + (size_t)j * (E_SZ + H_SZ);
        float p = 0.0f;
        for (int i = tid; i < E_SZ + H_SZ; i += 128) {
            float x = (i < H_SZ) ? h_new[i] : c_vec[i - H_SZ];
            p += row[i] * x;
        }
        p = block_reduce(p, red, tid, 128);
        if (tid == 0) out_vec[j] = tanhf(p + bo1[j]);
    }
}

// ---------------------------------------------------------------------------
// logits[r] = Wo2[r,:].out + bo2[r]   (Wo2 stays L2-resident across steps)
// ---------------------------------------------------------------------------
__global__ void logits_kernel(const float* __restrict__ Wo2,
                              const float* __restrict__ bo2,
                              const float* __restrict__ out_vec,
                              float* __restrict__ logits) {
    __shared__ float out_s[H_SZ];
    const int tid = threadIdx.x;
    out_s[tid] = out_vec[tid];
    out_s[tid + 256] = out_vec[tid + 256];
    __syncthreads();
    const int r = blockIdx.x * 256 + tid;
    if (r >= V_SZ) return;
    const float4* row = (const float4*)(Wo2 + (size_t)r * H_SZ);
    float acc = 0.0f;
    for (int k = 0; k < H_SZ / 4; ++k) {
        float4 w = row[k];
        acc += w.x * out_s[4 * k] + w.y * out_s[4 * k + 1] +
               w.z * out_s[4 * k + 2] + w.w * out_s[4 * k + 3];
    }
    logits[r] = acc + bo2[r];
}

// ---------------------------------------------------------------------------
// nll_acc += -(logits[y] - max - log(sum exp(logits - max)));  final write d_out
// ---------------------------------------------------------------------------
__global__ void nll_kernel(const float* __restrict__ logits,
                           const int* __restrict__ description,
                           float* __restrict__ nll_acc,
                           int t, float* __restrict__ d_out) {
    __shared__ float red[1024];
    const int tid = threadIdx.x;
    float mx = -3.4e38f;
    for (int i = tid; i < V_SZ; i += 1024) mx = fmaxf(mx, logits[i]);
    red[tid] = mx; __syncthreads();
    for (int s = 512; s > 0; s >>= 1) {
        if (tid < s) red[tid] = fmaxf(red[tid], red[tid + s]);
        __syncthreads();
    }
    mx = red[0]; __syncthreads();
    float sum = 0.0f;
    for (int i = tid; i < V_SZ; i += 1024) sum += expf(logits[i] - mx);
    red[tid] = sum; __syncthreads();
    for (int s = 512; s > 0; s >>= 1) {
        if (tid < s) red[tid] += red[tid + s];
        __syncthreads();
    }
    if (tid == 0) {
        const int y = description[t];
        const float lse = mx + logf(red[0]);
        float acc = *nll_acc + (lse - logits[y]);
        *nll_acc = acc;
        if (t == L_SZ - 1) d_out[0] = acc;
    }
}

// ---------------------------------------------------------------------------
extern "C" void kernel_launch(void* const* d_in, const int* in_sizes, int n_in,
                              void* d_out, int out_size, void* d_ws, size_t ws_size,
                              hipStream_t stream) {
    (void)in_sizes; (void)n_in; (void)out_size; (void)ws_size;
    const int*   context     = (const int*)  d_in[0];
    const int*   description = (const int*)  d_in[1];
    const float* emb_dec     = (const float*)d_in[2];
    const float* emb_ctx     = (const float*)d_in[3];
    const float* W1  = (const float*)d_in[4];
    const float* b1  = (const float*)d_in[5];
    const float* W2  = (const float*)d_in[6];
    const float* b2  = (const float*)d_in[7];
    const float* W3  = (const float*)d_in[8];
    const float* b3  = (const float*)d_in[9];
    const float* Wih = (const float*)d_in[10];
    const float* Whh = (const float*)d_in[11];
    const float* bih = (const float*)d_in[12];
    const float* bhh = (const float*)d_in[13];
    const float* Wm  = (const float*)d_in[14];
    const float* bm  = (const float*)d_in[15];
    const float* Wo1 = (const float*)d_in[16];
    const float* bo1 = (const float*)d_in[17];
    const float* Wo2 = (const float*)d_in[18];
    const float* bo2 = (const float*)d_in[19];

    char* ws = (char*)d_ws;
    size_t off = 0;
    auto alloc = [&](size_t bytes) {
        size_t o = off;
        off = (off + bytes + 255) & ~(size_t)255;
        return o;
    };
    float*  facts    = (float*) (ws + alloc(C_SZ * E_SZ * 4));
    bf16_t* Bpack    = (bf16_t*)(ws + alloc((size_t)K2_SZ * H_SZ * 2));
    bf16_t* zc       = (bf16_t*)(ws + alloc(C_SZ * K2_SZ * 2));
    float*  a_tanh   = (float*) (ws + alloc(C_SZ * H_SZ * 4));
    float*  h_buf0   = (float*) (ws + alloc(H_SZ * 4));
    float*  h_buf1   = (float*) (ws + alloc(H_SZ * 4));
    float*  m_buf0   = (float*) (ws + alloc(H_SZ * 4));
    float*  m_buf1   = (float*) (ws + alloc(H_SZ * 4));
    float*  gi       = (float*) (ws + alloc(3 * H_SZ * 4));
    float*  gh       = (float*) (ws + alloc(3 * H_SZ * 4));
    float*  c_vec    = (float*) (ws + alloc(H_SZ * 4));
    float*  out_vec  = (float*) (ws + alloc(H_SZ * 4));
    float*  logits   = (float*) (ws + alloc((size_t)V_SZ * 4));
    float*  nll_acc  = (float*) (ws + alloc(4));

    // ---- setup ----
    facts_kernel<<<C_SZ, 256, 0, stream>>>(context, emb_ctx, facts, nll_acc);
    packW1_kernel<<<1024, 32, 0, stream>>>(W1, Bpack);
    m0_kernel<<<H_SZ, 256, 0, stream>>>(Wm, bm, facts, m_buf0, h_buf0);

    float* hb[2] = { h_buf0, h_buf1 };
    float* mb[2] = { m_buf0, m_buf1 };

    // ---- sequential scan, unrolled into the launch stream ----
    for (int t = 0; t < L_SZ; ++t) {
        float* h_old = hb[t & 1];
        float* h_new = hb[(t + 1) & 1];
        float* m_old = mb[t & 1];
        float* m_new = mb[(t + 1) & 1];

        gru_matvec_kernel<<<3 * H_SZ, 128, 0, stream>>>(
            emb_dec, description, Wih, Whh, bih, bhh, m_old, h_old, gi, gh, t);
        gru_combine_kernel<<<2, 256, 0, stream>>>(gi, gh, h_old, h_new);
        zc_kernel<<<C_SZ, 256, 0, stream>>>(facts, h_new, m_old, zc);
        attn_wmma_kernel<<<dim3(4, 32), 32, 0, stream>>>(zc, Bpack, b1, a_tanh);
        score_ctx_kernel<<<1, 256, 0, stream>>>(a_tanh, W2, b2, facts, c_vec);
        memout_kernel<<<1024, 128, 0, stream>>>(
            W3, b3, Wo1, bo1, m_old, c_vec, h_new, m_new, out_vec);
        logits_kernel<<<(V_SZ + 255) / 256, 256, 0, stream>>>(Wo2, bo2, out_vec, logits);
        nll_kernel<<<1, 1024, 0, stream>>>(logits, description, nll_acc, t, (float*)d_out);
    }
}
